// MHA_63118839382398
// MI455X (gfx1250) — matrix operand order repaired and checked by reading, not compiled, not run on hardware
//
#include <hip/hip_runtime.h>

// ---------------------------------------------------------------------------
// CDNA5 (gfx1250) attention block: QKV proj + RMSNorm + RoPE + causal flash
// attention + out proj.  bf16 WMMA everywhere, TDM (tensor_load_to_lds) for
// K/V staging in the attention kernel, register-blocked GEMMs.
// ---------------------------------------------------------------------------

typedef __attribute__((ext_vector_type(16))) __bf16 v16bf;
typedef __attribute__((ext_vector_type(8)))  __bf16 v8bf;
typedef __attribute__((ext_vector_type(8)))  float  v8f;
typedef __attribute__((ext_vector_type(4)))  unsigned int ui4;
typedef __attribute__((ext_vector_type(8)))  unsigned int ui8;

#define WMMA_BF16(a, b, c) \
  __builtin_amdgcn_wmma_f32_16x16x32_bf16(false, (a), false, (b), (short)0, (c), false, false)

#define DMODEL 1024
#define NHEADS 16
#define DHEAD  64
#define SEQ    2048
#define BATCH  4
#define EQKV   3072
#define NTOK   (BATCH * SEQ)
#define RMS_EPS 1.1920928955078125e-07f
#define NEG_BIG -3.0e38f

// ---- fragment loaders (contiguous k = 16*(lane/16)+i in A and B, wave32) -------------------------
// A 16x32 bf16: lane row = lane%16; elems i = A[row][koff+i],
//               koff = (lane/16)*16.
__device__ __forceinline__ v16bf ldA(const __bf16* row, int koff) {
  v8bf a = *(const v8bf*)(row + koff);
  v8bf b = *(const v8bf*)(row + 8 + koff);
  return __builtin_shufflevector(a, b, 0,1,2,3,4,5,6,7,8,9,10,11,12,13,14,15);
}
// B 32x16 bf16: lane col = lane%16; elems i = B[kb+i][col], kb = (lane/16)*16.
// With K contiguous in memory for a fixed column this is one 32B load.
__device__ __forceinline__ v16bf ld16(const __bf16* p) {
  v8bf a = *(const v8bf*)(p);
  v8bf b = *(const v8bf*)(p + 8);
  return __builtin_shufflevector(a, b, 0,1,2,3,4,5,6,7,8,9,10,11,12,13,14,15);
}

__device__ __forceinline__ float redmax16(float v) {
  v = fmaxf(v, __shfl_xor(v, 1));
  v = fmaxf(v, __shfl_xor(v, 2));
  v = fmaxf(v, __shfl_xor(v, 4));
  v = fmaxf(v, __shfl_xor(v, 8));
  return v;
}
__device__ __forceinline__ float redsum16(float v) {
  v += __shfl_xor(v, 1);
  v += __shfl_xor(v, 2);
  v += __shfl_xor(v, 4);
  v += __shfl_xor(v, 8);
  return v;
}

// ---------------------------------------------------------------------------
// Tensor Data Mover: async 2-D bf16 tile load Global -> LDS.
// D# built per ISA 8.3-8.5: group0 {count=1, lds_addr, global_addr, type=2},
// group1 {data_size=2B, tensor_dim0/1, tile_dim0/1, dim0-stride}.  Groups 2/3
// are zero (tile_dim2 = 0 => dims >= 2 unused).  Tracked by TENSORcnt.
// ---------------------------------------------------------------------------
__device__ __forceinline__ void tdm_load_2d(unsigned lds_byte_addr,
                                            const void* gptr,
                                            unsigned tile0, unsigned tile1,
                                            unsigned stride0,
                                            unsigned td0, unsigned td1) {
  unsigned long long ga = (unsigned long long)(uintptr_t)gptr;
  ui4 g0;
  g0.x = 1u;                                              // count=1, load
  g0.y = lds_byte_addr;                                   // LDS dest (bytes)
  g0.z = (unsigned)ga;                                    // global_addr lo
  g0.w = ((unsigned)(ga >> 32) & 0x01FFFFFFu) | 0x80000000u;  // hi | type=2
  ui8 g1;
  g1[0] = 1u << 16;                                       // data_size = 2B
  g1[1] = (td0 & 0xFFFFu) << 16;                          // tensor_dim0 lo
  g1[2] = (td0 >> 16) | ((td1 & 0xFFFFu) << 16);          // dim0 hi | dim1 lo
  g1[3] = (td1 >> 16) | (tile0 << 16);                    // dim1 hi | tile0
  g1[4] = tile1;                                          // tile1 | tile2=0
  g1[5] = stride0;                                        // dim0 stride lo
  g1[6] = 0u;
  g1[7] = 0u;
  ui4 z = {0u, 0u, 0u, 0u};
  asm volatile("tensor_load_to_lds %0, %1, %2, %3"
               :: "s"(g0), "s"(g1), "s"(z), "s"(z) : "memory");
}

// ---------------------------------------------------------------------------
// fp32 -> bf16 conversion
// ---------------------------------------------------------------------------
__global__ void cvt_f32_bf16(const float* __restrict__ in,
                             __bf16* __restrict__ out, int n4) {
  int i = blockIdx.x * blockDim.x + threadIdx.x;
  if (i < n4) {
    float4 f = ((const float4*)in)[i];
    int j = i * 4;
    out[j + 0] = (__bf16)f.x;
    out[j + 1] = (__bf16)f.y;
    out[j + 2] = (__bf16)f.z;
    out[j + 3] = (__bf16)f.w;
  }
}

// ---------------------------------------------------------------------------
// QKV GEMM, register-blocked 2x4 tiles (32x64 output per wave):
// 8 WMMA per K-step against 12 b128 loads.  Epilogue scatters Q/K (B,H,N,d)
// and V transposed (B,H,d,N) in bf16.
// ---------------------------------------------------------------------------
__global__ __launch_bounds__(32)
void qkv_gemm(const __bf16* __restrict__ X, const __bf16* __restrict__ W,
              __bf16* __restrict__ Q, __bf16* __restrict__ K,
              __bf16* __restrict__ V) {
  const int lane = threadIdx.x;
  const int mt = blockIdx.x * 2;      // row tiles  (512 total)
  const int et = blockIdx.y * 4;      // col tiles  (192 total)
  const int l16  = lane & 15;
  const int hi   = lane >> 4;
  const int koff = hi * 16;
  const int kb   = hi * 16;

  const __bf16* xrow0 = X + (size_t)((mt + 0) * 16 + l16) * DMODEL;
  const __bf16* xrow1 = X + (size_t)((mt + 1) * 16 + l16) * DMODEL;
  const __bf16* wrow[4];
#pragma unroll
  for (int j = 0; j < 4; ++j)
    wrow[j] = W + (size_t)((et + j) * 16 + l16) * DMODEL;

  v8f acc[2][4] = {};
  for (int k0 = 0; k0 < DMODEL; k0 += 32) {
    v16bf a0 = ldA(xrow0, k0 + koff);
    v16bf a1 = ldA(xrow1, k0 + koff);
#pragma unroll
    for (int j = 0; j < 4; ++j) {
      v16bf b = ld16(wrow[j] + k0 + kb);
      acc[0][j] = WMMA_BF16(a0, b, acc[0][j]);
      acc[1][j] = WMMA_BF16(a1, b, acc[1][j]);
    }
  }

#pragma unroll
  for (int j = 0; j < 4; ++j) {
    const int e  = (et + j) * 16 + l16;
    const int h  = e / (3 * DHEAD);
    const int jj = e % (3 * DHEAD);
#pragma unroll
    for (int i = 0; i < 2; ++i) {
#pragma unroll
      for (int r = 0; r < 8; ++r) {
        const int M  = (mt + i) * 16 + hi * 8 + r;
        const int bb = M >> 11;
        const int n  = M & (SEQ - 1);
        const __bf16 val = (__bf16)acc[i][j][r];
        const size_t bh = (size_t)(bb * NHEADS + h);
        if (jj < DHEAD) {
          Q[(bh * SEQ + n) * DHEAD + jj] = val;
        } else if (jj < 2 * DHEAD) {
          K[(bh * SEQ + n) * DHEAD + (jj - DHEAD)] = val;
        } else {
          V[(bh * DHEAD + (jj - 2 * DHEAD)) * SEQ + n] = val;  // transposed
        }
      }
    }
  }
}

// ---------------------------------------------------------------------------
// RMSNorm (per head) + half-split RoPE, in place on Q and K.
// ---------------------------------------------------------------------------
__global__ __launch_bounds__(128)
void norm_rope(__bf16* __restrict__ Q, __bf16* __restrict__ K,
               const int* __restrict__ pos, const float* __restrict__ qw,
               const float* __restrict__ kw) {
  const int lane = threadIdx.x & 31;
  const int wave = threadIdx.x >> 5;
  const long item = (long)blockIdx.x * 4 + wave;   // (b*H + h)*SEQ + n
  const int n = (int)(item & (SEQ - 1));

  const float p = (float)pos[n];
  const float inv = __expf(-(float)lane * (9.210340371976184f / 32.0f));
  float s, c;
  __sincosf(p * inv, &s, &c);

  {
    __bf16* qp = Q + item * DHEAD;
    float t1 = (float)qp[lane];
    float t2 = (float)qp[lane + 32];
    float ss = t1 * t1 + t2 * t2;
    ss += __shfl_xor(ss, 1);  ss += __shfl_xor(ss, 2);
    ss += __shfl_xor(ss, 4);  ss += __shfl_xor(ss, 8);
    ss += __shfl_xor(ss, 16);
    const float r = rsqrtf(ss * (1.0f / 64.0f) + RMS_EPS);
    t1 *= r * qw[lane];
    t2 *= r * qw[lane + 32];
    qp[lane]      = (__bf16)(t1 * c - t2 * s);
    qp[lane + 32] = (__bf16)(t1 * s + t2 * c);
  }
  {
    __bf16* kp = K + item * DHEAD;
    float t1 = (float)kp[lane];
    float t2 = (float)kp[lane + 32];
    float ss = t1 * t1 + t2 * t2;
    ss += __shfl_xor(ss, 1);  ss += __shfl_xor(ss, 2);
    ss += __shfl_xor(ss, 4);  ss += __shfl_xor(ss, 8);
    ss += __shfl_xor(ss, 16);
    const float r = rsqrtf(ss * (1.0f / 64.0f) + RMS_EPS);
    t1 *= r * kw[lane];
    t2 *= r * kw[lane + 32];
    kp[lane]      = (__bf16)(t1 * c - t2 * s);
    kp[lane + 32] = (__bf16)(t1 * s + t2 * c);
  }
}

// ---------------------------------------------------------------------------
// Causal flash attention, TDM double-buffered.  One wave per 16-query tile.
// Per 32-key chunk: TDM stages K (32x64) and V^T (64x32) tiles into LDS for
// the NEXT chunk while this chunk computes S (4 WMMA), online softmax (fp32),
// P via LDS relayout, PV (4 WMMA).
// ---------------------------------------------------------------------------
__global__ __launch_bounds__(32)
void flash_attn(const __bf16* __restrict__ Q, const __bf16* __restrict__ K,
                const __bf16* __restrict__ V, __bf16* __restrict__ Y) {
  __shared__ __bf16 Kl[2][32 * 64];   // [key][d]
  __shared__ __bf16 Vl[2][64 * 32];   // [d][key]
  __shared__ __bf16 Plds[16 * 32];

  const int lane = threadIdx.x;
  const int qt = blockIdx.x;          // query tile 0..127
  const int bh = blockIdx.y;          // b*H + h, 0..63
  const int l16 = lane & 15;
  const int hi  = lane >> 4;
  const int koff = hi * 16;
  const int kb16 = hi * 16;

  const __bf16* Qb = Q + (size_t)bh * SEQ * DHEAD;
  const __bf16* Kb = K + (size_t)bh * SEQ * DHEAD;
  const __bf16* Vb = V + (size_t)bh * DHEAD * SEQ;

  const __bf16* qrow = Qb + (size_t)(qt * 16 + l16) * DHEAD;
  const v16bf qa0 = ldA(qrow, koff);
  const v16bf qa1 = ldA(qrow + 32, koff);

  float m[8], l[8];
  v8f acc[4] = {};
  for (int r = 0; r < 8; ++r) { m[r] = NEG_BIG; l[r] = 0.0f; }

  const int qbase = qt * 16 + hi * 8;
  const int kcEnd = (qt * 16 + 15) >> 5;

  // Prologue: TDM-stage chunk 0 into buffer 0.
  tdm_load_2d((unsigned)(uintptr_t)&Kl[0][0], Kb,
              /*tile0=*/DHEAD, /*tile1=*/32, /*stride0=*/DHEAD,
              /*td0=*/DHEAD, /*td1=*/SEQ);
  tdm_load_2d((unsigned)(uintptr_t)&Vl[0][0], Vb,
              /*tile0=*/32, /*tile1=*/DHEAD, /*stride0=*/SEQ,
              /*td0=*/SEQ, /*td1=*/DHEAD);

  for (int kc = 0; kc <= kcEnd; ++kc) {
    const int kbase = kc * 32;
    const int buf = kc & 1;

    // Stage next chunk into the other buffer, then wait for this chunk.
    if (kc < kcEnd) {
      const int nb = buf ^ 1;
      tdm_load_2d((unsigned)(uintptr_t)&Kl[nb][0],
                  Kb + (size_t)(kbase + 32) * DHEAD,
                  DHEAD, 32, DHEAD, DHEAD, SEQ);
      tdm_load_2d((unsigned)(uintptr_t)&Vl[nb][0],
                  Vb + (size_t)(kbase + 32),
                  32, DHEAD, SEQ, SEQ, DHEAD);
      __builtin_amdgcn_s_wait_tensorcnt(2);   // current chunk landed
    } else {
      __builtin_amdgcn_s_wait_tensorcnt(0);
    }

    // ---- S = Q * K^T for 32 keys (two 16x16 tiles, B frags from LDS) ----
    const __bf16* k0 = &Kl[buf][(size_t)l16 * DHEAD];
    const __bf16* k1 = &Kl[buf][(size_t)(16 + l16) * DHEAD];
    v8f s0 = {}, s1 = {};
    s0 = WMMA_BF16(qa0, ld16(k0 + kb16), s0);
    s0 = WMMA_BF16(qa1, ld16(k0 + 32 + kb16), s0);
    s1 = WMMA_BF16(qa0, ld16(k1 + kb16), s1);
    s1 = WMMA_BF16(qa1, ld16(k1 + 32 + kb16), s1);

    // ---- online softmax ----
    const int key0 = kbase + l16;
    const int key1 = key0 + 16;
#pragma unroll
    for (int r = 0; r < 8; ++r) {
      const int qr = qbase + r;
      float v0 = (key0 <= qr) ? s0[r] * 0.125f : NEG_BIG;
      float v1 = (key1 <= qr) ? s1[r] * 0.125f : NEG_BIG;
      const float mx = redmax16(fmaxf(v0, v1));
      const float mn = fmaxf(m[r], mx);
      const float alpha = __expf(m[r] - mn);
      const float p0 = __expf(v0 - mn);
      const float p1 = __expf(v1 - mn);
      l[r] = l[r] * alpha + redsum16(p0 + p1);
      m[r] = mn;
      s0[r] = p0;
      s1[r] = p1;
      acc[0][r] *= alpha; acc[1][r] *= alpha;
      acc[2][r] *= alpha; acc[3][r] *= alpha;
    }

    // ---- P: C-layout -> LDS 16x32 -> A-layout fragment ----
#pragma unroll
    for (int r = 0; r < 8; ++r) {
      Plds[(hi * 8 + r) * 32 + l16]      = (__bf16)s0[r];
      Plds[(hi * 8 + r) * 32 + l16 + 16] = (__bf16)s1[r];
    }
    __syncthreads();
    const v16bf pa = ldA(&Plds[l16 * 32], koff);
    __syncthreads();

    // ---- acc += P * V  (V tile already [d][key] in LDS) ----
#pragma unroll
    for (int t = 0; t < 4; ++t) {
      const __bf16* vrow = &Vl[buf][(size_t)(t * 16 + l16) * 32 + kb16];
      acc[t] = WMMA_BF16(pa, ld16(vrow), acc[t]);
    }
  }

  // ---- epilogue ----
  const int b = bh >> 4;
  const int h = bh & 15;
#pragma unroll
  for (int r = 0; r < 8; ++r) {
    const float inv_l = 1.0f / l[r];
    const int tok = b * SEQ + qbase + r;
#pragma unroll
    for (int t = 0; t < 4; ++t) {
      Y[(size_t)tok * DMODEL + h * DHEAD + t * 16 + l16] =
          (__bf16)(acc[t][r] * inv_l);
    }
  }
}

// ---------------------------------------------------------------------------
// Output GEMM, register-blocked 2x4 tiles, fp32 result.
// ---------------------------------------------------------------------------
__global__ __launch_bounds__(32)
void out_gemm(const __bf16* __restrict__ Yb, const __bf16* __restrict__ W,
              float* __restrict__ out) {
  const int lane = threadIdx.x;
  const int mt = blockIdx.x * 2;
  const int et = blockIdx.y * 4;
  const int l16  = lane & 15;
  const int hi   = lane >> 4;
  const int koff = hi * 16;
  const int kb   = hi * 16;

  const __bf16* yrow0 = Yb + (size_t)((mt + 0) * 16 + l16) * DMODEL;
  const __bf16* yrow1 = Yb + (size_t)((mt + 1) * 16 + l16) * DMODEL;
  const __bf16* wrow[4];
#pragma unroll
  for (int j = 0; j < 4; ++j)
    wrow[j] = W + (size_t)((et + j) * 16 + l16) * DMODEL;

  v8f acc[2][4] = {};
  for (int k0 = 0; k0 < DMODEL; k0 += 32) {
    v16bf a0 = ldA(yrow0, k0 + koff);
    v16bf a1 = ldA(yrow1, k0 + koff);
#pragma unroll
    for (int j = 0; j < 4; ++j) {
      v16bf b = ld16(wrow[j] + k0 + kb);
      acc[0][j] = WMMA_BF16(a0, b, acc[0][j]);
      acc[1][j] = WMMA_BF16(a1, b, acc[1][j]);
    }
  }

#pragma unroll
  for (int j = 0; j < 4; ++j) {
    const int col = (et + j) * 16 + l16;
#pragma unroll
    for (int i = 0; i < 2; ++i) {
#pragma unroll
      for (int r = 0; r < 8; ++r) {
        const int M = (mt + i) * 16 + hi * 8 + r;
        out[(size_t)M * DMODEL + col] = acc[i][j][r];
      }
    }
  }
}

// ---------------------------------------------------------------------------
// Host launcher
// ---------------------------------------------------------------------------
extern "C" void kernel_launch(void* const* d_in, const int* in_sizes, int n_in,
                              void* d_out, int out_size, void* d_ws, size_t ws_size,
                              hipStream_t stream) {
  const float* x     = (const float*)d_in[0];
  // d_in[1] = mask (bool) — causality applied analytically in-kernel.
  const int*   pos   = (const int*)d_in[2];
  const float* W_qkv = (const float*)d_in[3];
  const float* W_out = (const float*)d_in[4];
  const float* qn_w  = (const float*)d_in[5];
  const float* kn_w  = (const float*)d_in[6];
  float* out = (float*)d_out;

  char* ws = (char*)d_ws;
  __bf16* xb     = (__bf16*)(ws);                      // 16 MB
  __bf16* Wqkv_b = (__bf16*)(ws + 16777216);           //  6 MB
  __bf16* Wout_b = (__bf16*)(ws + 23068672);           //  2 MB
  __bf16* Qbuf   = (__bf16*)(ws + 25165824);           // 16 MB (B,H,N,d)
  __bf16* Kbuf   = (__bf16*)(ws + 41943040);           // 16 MB (B,H,N,d)
  __bf16* Vt     = (__bf16*)(ws + 58720256);           // 16 MB (B,H,d,N)
  __bf16* Yb     = (__bf16*)(ws + 75497472);           // 16 MB (B*N, D)

  {
    int n4 = (NTOK * DMODEL) / 4;
    cvt_f32_bf16<<<(n4 + 255) / 256, 256, 0, stream>>>(x, xb, n4);
    n4 = (EQKV * DMODEL) / 4;
    cvt_f32_bf16<<<(n4 + 255) / 256, 256, 0, stream>>>(W_qkv, Wqkv_b, n4);
    n4 = (DMODEL * DMODEL) / 4;
    cvt_f32_bf16<<<(n4 + 255) / 256, 256, 0, stream>>>(W_out, Wout_b, n4);
  }

  qkv_gemm<<<dim3(NTOK / 32, EQKV / 64), 32, 0, stream>>>(xb, Wqkv_b,
                                                          Qbuf, Kbuf, Vt);

  norm_rope<<<(BATCH * NHEADS * SEQ) / 4, 128, 0, stream>>>(Qbuf, Kbuf, pos,
                                                            qn_w, kn_w);

  flash_attn<<<dim3(SEQ / 16, BATCH * NHEADS), 32, 0, stream>>>(Qbuf, Kbuf,
                                                                Vt, Yb);

  out_gemm<<<dim3(NTOK / 32, DMODEL / 64), 32, 0, stream>>>(Yb, Wout_b, out);
}
